// GATLayer_77627238908015
// MI455X (gfx1250) — matrix operand order repaired and checked by reading, not compiled, not run on hardware
//
#include <hip/hip_runtime.h>
#include <hip/hip_bf16.h>
#include <math.h>

#define H      8
#define HD     8
#define DIN    64
#define DOUT   64
#define NEG_SLOPE 0.2f

typedef __attribute__((ext_vector_type(16))) _Float16 v16h;
typedef __attribute__((ext_vector_type(8)))  float    v8f;

// ---------------------------------------------------------------------------
// A-operand loader: 16-bit A 16x32 layout (ISA 7.12.2):
//   lanes 0-15  : row M = lane,   V0..V3 = K 0..7,  V4..V7 = K 16..23
//   lanes 16-31 : row M = lane-16, same pattern shifted by +8 in K
// p points at (row base + ktile*32 + khalfA); we pull K offsets {0..7,16..23},
// splitting each f32 into f16 hi + f16 lo for the 3-WMMA f32-emulation.
// ---------------------------------------------------------------------------
__device__ __forceinline__ void load_splitA(const float* __restrict__ p,
                                            v16h& hi, v16h& lo) {
  float4 f0 = *reinterpret_cast<const float4*>(p + 0);
  float4 f1 = *reinterpret_cast<const float4*>(p + 4);
  float4 f2 = *reinterpret_cast<const float4*>(p + 16);
  float4 f3 = *reinterpret_cast<const float4*>(p + 20);
  float v[16] = {f0.x,f0.y,f0.z,f0.w, f1.x,f1.y,f1.z,f1.w,
                 f2.x,f2.y,f2.z,f2.w, f3.x,f3.y,f3.z,f3.w};
#pragma unroll
  for (int i = 0; i < 16; ++i) {
    _Float16 h = (_Float16)v[i];
    hi[i] = h;
    lo[i] = (_Float16)(v[i] - (float)h);
  }
}

// ---------------------------------------------------------------------------
// B-operand loader: 16-bit B 32x16 layout matching A: lane = column N;
// lanes 0-15 hold K=0..7,16..23, lanes 16-31 hold K=8..15,24..31.
// p points at (col-row base + ktile*32 + kbaseB); K offsets {0..7,16..23}.
// ---------------------------------------------------------------------------
__device__ __forceinline__ void load_splitB(const float* __restrict__ p,
                                            v16h& hi, v16h& lo) {
  float4 f0 = *reinterpret_cast<const float4*>(p + 0);
  float4 f1 = *reinterpret_cast<const float4*>(p + 4);
  float4 f2 = *reinterpret_cast<const float4*>(p + 16);
  float4 f3 = *reinterpret_cast<const float4*>(p + 20);
  float v[16] = {f0.x,f0.y,f0.z,f0.w, f1.x,f1.y,f1.z,f1.w,
                 f2.x,f2.y,f2.z,f2.w, f3.x,f3.y,f3.z,f3.w};
#pragma unroll
  for (int i = 0; i < 16; ++i) {
    _Float16 h = (_Float16)v[i];
    hi[i] = h;
    lo[i] = (_Float16)(v[i] - (float)h);
  }
}

// ---------------------------------------------------------------------------
// Kernel 1: xp = x @ W^T via WMMA. One wave32 computes a 16x64 output tile.
// f32 accuracy recovered with hi/lo f16 split (3 WMMAs per K-tile).
// ---------------------------------------------------------------------------
__global__ void gat_proj_wmma(const float* __restrict__ x,
                              const float* __restrict__ W,
                              float* __restrict__ xp,
                              int n_nodes, int n_tiles) {
  const int wave = (blockIdx.x * blockDim.x + threadIdx.x) >> 5;
  const int lane = threadIdx.x & 31;
  if (wave >= n_tiles) return;

  const int row0   = wave * 16;
  const int r      = lane & 15;
  const bool hiLane = (lane >= 16);
  const int khalfA = hiLane ? 8 : 0;   // A K-offset pattern
  const int kbaseB = hiLane ? 8 : 0;   // B K-offset pattern

  int arow = row0 + r;
  if (arow >= n_nodes) arow = n_nodes - 1;  // safe clamp (N % 16 == 0 anyway)

  // Load both K-tiles of A once (x row is 64 floats).
  v16h ahi[2], alo[2];
#pragma unroll
  for (int kt = 0; kt < 2; ++kt)
    load_splitA(x + (size_t)arow * DIN + kt * 32 + khalfA, ahi[kt], alo[kt]);

#pragma unroll
  for (int nt = 0; nt < 4; ++nt) {
    v8f acc = {};
#pragma unroll
    for (int kt = 0; kt < 2; ++kt) {
      // B[k][n] = W^T[k][n] = W[n][k] -> lane n reads W row (nt*16 + r)
      v16h bhi, blo;
      load_splitB(W + (size_t)(nt * 16 + r) * DIN + kt * 32 + kbaseB, bhi, blo);
      // acc += lo*hi + hi*lo + hi*hi  (f32 accumulate inside WMMA)
      acc = __builtin_amdgcn_wmma_f32_16x16x32_f16(false, alo[kt], false, bhi,
                                                   (short)0, acc, false, false);
      acc = __builtin_amdgcn_wmma_f32_16x16x32_f16(false, ahi[kt], false, blo,
                                                   (short)0, acc, false, false);
      acc = __builtin_amdgcn_wmma_f32_16x16x32_f16(false, ahi[kt], false, bhi,
                                                   (short)0, acc, false, false);
    }
    // C/D layout: lane (0-15)=col n, VGPR m = row m; lanes 16-31 rows +8.
    const int col   = nt * 16 + r;
    const int rbase = row0 + (hiLane ? 8 : 0);
#pragma unroll
    for (int m = 0; m < 8; ++m) {
      const int row = rbase + m;
      if (row < n_nodes) xp[(size_t)row * DOUT + col] = acc[m];
    }
  }
}

// ---------------------------------------------------------------------------
// Kernel 2: per-(node,head) attention logits + init of segment max / denom.
// src_score[i] = <xp[i*8..], src_attn[h]> ; likewise dst. i = node*H + h.
// ---------------------------------------------------------------------------
__global__ void gat_scores(const float* __restrict__ xp,
                           const float* __restrict__ src_attn,
                           const float* __restrict__ dst_attn,
                           float* __restrict__ src_score,
                           float* __restrict__ dst_score,
                           float* __restrict__ seg_max,
                           float* __restrict__ seg_den,
                           int n_nodes) {
  const int idx = blockIdx.x * blockDim.x + threadIdx.x;  // node*H + h
  if (idx >= n_nodes * H) return;
  const int h = idx & (H - 1);
  const float* v = xp + (size_t)idx * HD;  // (node*H + h)*HD
  float4 a0 = *reinterpret_cast<const float4*>(v);
  float4 a1 = *reinterpret_cast<const float4*>(v + 4);
  float4 s0 = *reinterpret_cast<const float4*>(src_attn + h * HD);
  float4 s1 = *reinterpret_cast<const float4*>(src_attn + h * HD + 4);
  float4 d0 = *reinterpret_cast<const float4*>(dst_attn + h * HD);
  float4 d1 = *reinterpret_cast<const float4*>(dst_attn + h * HD + 4);
  float ss = a0.x*s0.x + a0.y*s0.y + a0.z*s0.z + a0.w*s0.w +
             a1.x*s1.x + a1.y*s1.y + a1.z*s1.z + a1.w*s1.w;
  float dd = a0.x*d0.x + a0.y*d0.y + a0.z*d0.z + a0.w*d0.w +
             a1.x*d1.x + a1.y*d1.y + a1.z*d1.z + a1.w*d1.w;
  src_score[idx] = ss;
  dst_score[idx] = dd;
  seg_max[idx] = -INFINITY;
  seg_den[idx] = 0.0f;
}

__global__ void gat_zero(float* __restrict__ p, int n) {
  const int i = blockIdx.x * blockDim.x + threadIdx.x;
  if (i < n) p[i] = 0.0f;
}

__device__ __forceinline__ float edge_logit(const float* __restrict__ ss,
                                            const float* __restrict__ ds,
                                            int row, int col, int h) {
  const float e = ss[row * H + h] + ds[col * H + h];
  return e >= 0.0f ? e : NEG_SLOPE * e;
}

// Float atomic max via int/uint punning (exact; needs init = -inf).
__device__ __forceinline__ void atomicMaxFloat(float* addr, float v) {
  if (v >= 0.0f) atomicMax(reinterpret_cast<int*>(addr), __float_as_int(v));
  else           atomicMin(reinterpret_cast<unsigned int*>(addr), __float_as_uint(v));
}

// ---------------------------------------------------------------------------
// Kernel 3: segment max over rows.  One thread per (edge, head).
// ---------------------------------------------------------------------------
__global__ void gat_edge_max(const int* __restrict__ ei,
                             const float* __restrict__ ss,
                             const float* __restrict__ ds,
                             float* __restrict__ seg_max,
                             int n_edges) {
  const int t = blockIdx.x * blockDim.x + threadIdx.x;
  if (t >= n_edges * H) return;
  const int e = t >> 3, h = t & (H - 1);
  const int row = ei[e];
  const int col = ei[n_edges + e];
  atomicMaxFloat(seg_max + row * H + h, edge_logit(ss, ds, row, col, h));
}

// ---------------------------------------------------------------------------
// Kernel 4: segment sum of exp(e - max).  Logit recomputed (cheaper than
// materializing 40 MB of per-edge data).
// ---------------------------------------------------------------------------
__global__ void gat_edge_sum(const int* __restrict__ ei,
                             const float* __restrict__ ss,
                             const float* __restrict__ ds,
                             const float* __restrict__ seg_max,
                             float* __restrict__ seg_den,
                             int n_edges) {
  const int t = blockIdx.x * blockDim.x + threadIdx.x;
  if (t >= n_edges * H) return;
  const int e = t >> 3, h = t & (H - 1);
  const int row = ei[e];
  const int col = ei[n_edges + e];
  const int sh = row * H + h;
  const float ex = __expf(edge_logit(ss, ds, row, col, h) - seg_max[sh]);
  atomicAdd(seg_den + sh, ex);
}

// ---------------------------------------------------------------------------
// Kernel 5: out[row] += alpha * xp[col].  One thread per (edge, head),
// each thread scatter-adds its 8-float head slice. xp fits in L2 (25.6 MB).
// ---------------------------------------------------------------------------
__global__ void gat_aggregate(const int* __restrict__ ei,
                              const float* __restrict__ ss,
                              const float* __restrict__ ds,
                              const float* __restrict__ seg_max,
                              const float* __restrict__ seg_den,
                              const float* __restrict__ xp,
                              float* __restrict__ out,
                              int n_edges) {
  const int t = blockIdx.x * blockDim.x + threadIdx.x;
  if (t >= n_edges * H) return;
  const int e = t >> 3, h = t & (H - 1);
  const int row = ei[e];
  const int col = ei[n_edges + e];
  const int sh = row * H + h;
  const float alpha = __expf(edge_logit(ss, ds, row, col, h) - seg_max[sh])
                      / seg_den[sh];
  const float* src = xp + (size_t)col * DOUT + h * HD;
  float4 f0 = *reinterpret_cast<const float4*>(src);
  float4 f1 = *reinterpret_cast<const float4*>(src + 4);
  float* dst = out + (size_t)row * DOUT + h * HD;
  atomicAdd(dst + 0, alpha * f0.x);
  atomicAdd(dst + 1, alpha * f0.y);
  atomicAdd(dst + 2, alpha * f0.z);
  atomicAdd(dst + 3, alpha * f0.w);
  atomicAdd(dst + 4, alpha * f1.x);
  atomicAdd(dst + 5, alpha * f1.y);
  atomicAdd(dst + 6, alpha * f1.z);
  atomicAdd(dst + 7, alpha * f1.w);
}

// ---------------------------------------------------------------------------
// Launch. Inputs: x [N,64] f32, edge_indices [2,E] i32, W [64,64] f32,
// src_attn [1,8,8] f32, dst_attn [1,8,8] f32.  Output: [N,64] f32.
// Workspace: xp (N*64) + src/dst scores (N*8 each) + max/den (N*8 each)
//          = 38.4 MB for N = 100k.
// ---------------------------------------------------------------------------
extern "C" void kernel_launch(void* const* d_in, const int* in_sizes, int n_in,
                              void* d_out, int out_size, void* d_ws, size_t ws_size,
                              hipStream_t stream) {
  const float* x        = (const float*)d_in[0];
  const int*   ei       = (const int*)  d_in[1];
  const float* W        = (const float*)d_in[2];
  const float* src_attn = (const float*)d_in[3];
  const float* dst_attn = (const float*)d_in[4];
  float*       out      = (float*)d_out;

  const int n_nodes = in_sizes[0] / DIN;
  const int n_edges = in_sizes[1] / 2;

  char* ws = (char*)d_ws;
  float* xp      = (float*)ws; ws += (size_t)n_nodes * DOUT * sizeof(float);
  float* ss      = (float*)ws; ws += (size_t)n_nodes * H    * sizeof(float);
  float* ds      = (float*)ws; ws += (size_t)n_nodes * H    * sizeof(float);
  float* seg_max = (float*)ws; ws += (size_t)n_nodes * H    * sizeof(float);
  float* seg_den = (float*)ws;

  const int TB = 256;

  // 1) projection (WMMA)
  const int n_tiles = (n_nodes + 15) / 16;
  const int proj_threads = n_tiles * 32;
  gat_proj_wmma<<<(proj_threads + TB - 1) / TB, TB, 0, stream>>>(
      x, W, xp, n_nodes, n_tiles);

  // zero the output accumulator (d_out is poisoned by the harness)
  const int out_elems = n_nodes * DOUT;
  gat_zero<<<(out_elems + TB - 1) / TB, TB, 0, stream>>>(out, out_elems);

  // 2) per-node scores + init max/den
  const int nh = n_nodes * H;
  gat_scores<<<(nh + TB - 1) / TB, TB, 0, stream>>>(
      xp, src_attn, dst_attn, ss, ds, seg_max, seg_den, n_nodes);

  // 3/4/5) edge passes
  const int eh = n_edges * H;
  const int eblocks = (eh + TB - 1) / TB;
  gat_edge_max  <<<eblocks, TB, 0, stream>>>(ei, ss, ds, seg_max, n_edges);
  gat_edge_sum  <<<eblocks, TB, 0, stream>>>(ei, ss, ds, seg_max, seg_den, n_edges);
  gat_aggregate <<<eblocks, TB, 0, stream>>>(ei, ss, ds, seg_max, seg_den,
                                             xp, out, n_edges);
}
